// lpLSTM_3161095930312
// MI455X (gfx1250) — matrix operand order repaired and checked by reading, not compiled, not run on hardware
//
#include <hip/hip_runtime.h>

// ---------------------------------------------------------------------------
// lpLSTM for MI455X (gfx1250): bf16 WMMA GEMMs + persistent fused scan kernel.
//   T=512, B=64, I=H=1024, 4H=4096
// ---------------------------------------------------------------------------

#define T_STEPS 512
#define BATCH   64
#define HID     1024
#define G4      4096
#define KDIM    1024
#define NBLK    64          // blocks in the persistent scan kernel

typedef __attribute__((ext_vector_type(16))) __bf16 v16bf;
typedef __attribute__((ext_vector_type(8)))  float  v8f;
typedef __attribute__((ext_vector_type(4)))  int    v4i;

union FragBF {
  uint4 u[2];
  v16bf v;
};

// ---------------------------------------------------------------------------
// gfx1250 async global->LDS copy (ASYNCcnt-tracked). Parameter types per the
// compiler diagnostic: (AS1 v4i* global_src, AS3 v4i* lds_dst, imm, imm).
// Guarded so the file still compiles on toolchains without the builtin.
// ---------------------------------------------------------------------------
#if defined(__has_builtin)
# if __has_builtin(__builtin_amdgcn_global_load_async_to_lds_b128)
#  define HAVE_ASYNC_LDS 1
# endif
#endif
#ifndef HAVE_ASYNC_LDS
# define HAVE_ASYNC_LDS 0
#endif

__device__ __forceinline__ void async_cp16(void* l, const void* g) {
#if HAVE_ASYNC_LDS
  typedef __attribute__((address_space(1))) v4i g_v4i;
  typedef __attribute__((address_space(3))) v4i l_v4i;
  __builtin_amdgcn_global_load_async_to_lds_b128(
      (g_v4i*)(uintptr_t)g,
      (l_v4i*)(uint32_t)(uintptr_t)l,   // low 32 bits of generic LDS ptr
      0, 0);
#else
  *reinterpret_cast<uint4*>(l) = *reinterpret_cast<const uint4*>(g);
#endif
}

__device__ __forceinline__ void wait_async_all() {
#if HAVE_ASYNC_LDS
# if __has_builtin(__builtin_amdgcn_s_wait_asynccnt)
  __builtin_amdgcn_s_wait_asynccnt(0);
# else
  asm volatile("s_wait_asynccnt 0" ::: "memory");
# endif
#endif
}

// ---------------------------------------------------------------------------
// Fragment loaders
// ---------------------------------------------------------------------------
// A-matrix 16x32 bf16 fragment (row-major global, row length 1024):
// lanes 0-15 : K = k+0..7, k+16..23 ; lanes 16-31: K = k+8..15, k+24..31
__device__ __forceinline__ v16bf load_a_frag(const __bf16* __restrict__ base,
                                             size_t row, int k, int lane) {
  const __bf16* p = base + row * (size_t)KDIM + k + ((lane >> 4) << 3);
  FragBF f;
  f.u[0] = *reinterpret_cast<const uint4*>(p);
  f.u[1] = *reinterpret_cast<const uint4*>(p + 16);
  return f.v;
}

// LDS fragment loaders on 64-element rows padded to 9 uint4 (144 B):
// row starts land at banks 36*r mod 64 -> 16 distinct 4-bank spans, no conflicts.
#define LDS64_ROW_U4 9
__device__ __forceinline__ v16bf lds_a_frag64(const uint4* buf, int r, int kk, int lane) {
  const char* p = (const char*)(buf + (size_t)r * LDS64_ROW_U4) +
                  ((kk + ((lane >> 4) << 3)) << 1);
  FragBF f;
  f.u[0] = *reinterpret_cast<const uint4*>(p);
  f.u[1] = *reinterpret_cast<const uint4*>(p + 32);
  return f.v;
}
__device__ __forceinline__ v16bf lds_b_frag64(const uint4* buf, int r, int kk, int lane) {
  const char* p = (const char*)(buf + (size_t)r * LDS64_ROW_U4) +
                  ((kk + ((lane >> 4) << 3)) << 1);
  FragBF f;
  f.u[0] = *reinterpret_cast<const uint4*>(p);
  f.u[1] = *reinterpret_cast<const uint4*>(p + 32);
  return f.v;
}

// Scan-kernel LDS B fragment, 1024-element rows padded to 129 uint4 (2064 B).
#define LDS_ROW_U4 129
__device__ __forceinline__ v16bf lds_b_frag(const uint4* smem, int r, int k, int lane) {
  const char* p = (const char*)(smem + (size_t)r * LDS_ROW_U4) +
                  ((k + ((lane >> 4) << 3)) << 1);
  FragBF f;
  f.u[0] = *reinterpret_cast<const uint4*>(p);
  f.u[1] = *reinterpret_cast<const uint4*>(p + 32);
  return f.v;
}

__device__ __forceinline__ float sigf(float x) {
  return 1.0f / (1.0f + __expf(-x));
}

// ---------------------------------------------------------------------------
// fp32 -> bf16 bulk convert
// ---------------------------------------------------------------------------
__global__ void cvt_f32_bf16(const float* __restrict__ in,
                             __bf16* __restrict__ out, long long n) {
  long long i = (long long)blockIdx.x * blockDim.x + threadIdx.x;
  long long stride = (long long)gridDim.x * blockDim.x;
  for (; i < n; i += stride) out[i] = (__bf16)in[i];
}

// ---------------------------------------------------------------------------
// state init: h0 -> h_f32 & h_bf16, c0 -> c
// ---------------------------------------------------------------------------
__global__ void init_state(const float* __restrict__ h0, const float* __restrict__ c0,
                           float* __restrict__ hf, __bf16* __restrict__ hb,
                           float* __restrict__ cf) {
  int i = blockIdx.x * blockDim.x + threadIdx.x;
  float h = h0[i];
  hf[i] = h;
  hb[i] = (__bf16)h;
  cf[i] = c0[i];
}

// ---------------------------------------------------------------------------
// x_gates GEMM (LDS-tiled, double-buffered, async-staged):
//   XG[t*B+b, n] = sum_k X[t*B+b, k] * Wih[n, k] + bih[n] + bhh[n]
// M = 32768, N = 4096, K = 1024.
// Block (256 thr, 8 waves) -> 128x64 output tile; wave w owns 16 rows x 64
// cols (4 accumulators). K consumed in 64-wide chunks staged in LDS.
// ---------------------------------------------------------------------------
__global__ __launch_bounds__(256)
void xgates_gemm(const __bf16* __restrict__ X, const __bf16* __restrict__ Wih,
                 const float* __restrict__ bih, const float* __restrict__ bhh,
                 __bf16* __restrict__ XG) {
  __shared__ uint4 sA[2][128 * LDS64_ROW_U4];  // 2 x 18,432 B
  __shared__ uint4 sB[2][64 * LDS64_ROW_U4];   // 2 x  9,216 B

  const int tid  = threadIdx.x;
  const int lane = tid & 31;
  const int wave = tid >> 5;                 // 0..7 = m-subtile
  const size_t m0 = (size_t)blockIdx.x * 128;
  const int    n0 = blockIdx.y * 64;

  // stage one K-chunk (A: 1024 uint4, B: 512 uint4) into the given buffers
  auto stage = [&](int kc, uint4* aBuf, uint4* bBuf) {
#pragma unroll
    for (int i = 0; i < 4; ++i) {            // A: 4 uint4 per thread
      const int idx = (i << 8) + tid;        // 0..1023
      const int r = idx >> 3, c = idx & 7;
      async_cp16(aBuf + (size_t)r * LDS64_ROW_U4 + c,
                 X + (m0 + r) * KDIM + kc + (c << 3));
    }
#pragma unroll
    for (int i = 0; i < 2; ++i) {            // B: 2 uint4 per thread
      const int idx = (i << 8) + tid;        // 0..511
      const int r = idx >> 3, c = idx & 7;
      async_cp16(bBuf + (size_t)r * LDS64_ROW_U4 + c,
                 Wih + (size_t)(n0 + r) * KDIM + kc + (c << 3));
    }
  };

  v8f acc[4] = {v8f{}, v8f{}, v8f{}, v8f{}};
  const int arow = wave * 16 + (lane & 15);

  stage(0, sA[0], sB[0]);
  wait_async_all();
  __syncthreads();

  for (int kc = 0; kc < 16; ++kc) {          // 16 chunks of 64
    const int buf = kc & 1;
    if (kc + 1 < 16) stage((kc + 1) * 64, sA[buf ^ 1], sB[buf ^ 1]);

#pragma unroll
    for (int kk = 0; kk < 64; kk += 32) {
      v16bf a = lds_a_frag64(sA[buf], arow, kk, lane);
#pragma unroll
      for (int j = 0; j < 4; ++j) {
        v16bf b = lds_b_frag64(sB[buf], j * 16 + (lane & 15), kk, lane);
        acc[j] = __builtin_amdgcn_wmma_f32_16x16x32_bf16(false, a, false, b,
                                                         (short)0, acc[j], false, false);
      }
    }

    wait_async_all();      // our async stage of buf^1 complete
    __syncthreads();       // everyone done reading buf / writing buf^1
  }

  // epilogue: + (bias_ih + bias_hh), store bf16
  const int rbase = (int)m0 + wave * 16 + ((lane >> 4) << 3);
#pragma unroll
  for (int j = 0; j < 4; ++j) {
    const int col = n0 + j * 16 + (lane & 15);
    const float bias = bih[col] + bhh[col];
#pragma unroll
    for (int v = 0; v < 8; ++v) {
      XG[(size_t)(rbase + v) * G4 + col] = (__bf16)(acc[j][v] + bias);
    }
  }
}

// ---------------------------------------------------------------------------
// Persistent scan kernel. 64 blocks x 256 threads (8 waves).
// Block b owns hidden n-tile b (16 columns) for the entire batch.
// Wave w: m-tile = w>>1 (16 batch rows), k-half = w&1 (512 of K=1024).
// W_hh slice (64 gate-rows x 1024 K = 128KB bf16) staged in LDS once and
// reused for all 512 timesteps; per step the k1 waves dump partial gate
// accumulators to LDS, k0 waves reduce + do the elementwise LSTM/blend.
// Device-wide generation barrier (agent-scope atomics) between steps.
// ---------------------------------------------------------------------------
__global__ __launch_bounds__(256)
void lstm_scan(const __bf16* __restrict__ XG,   // [T, B, 4H]
               const __bf16* __restrict__ Whh,  // [4H, H]
               const float* __restrict__ ret,   // [H]
               float* h_f0, float* h_f1,        // [B, H] f32 ping-pong
               __bf16* h_b0, __bf16* h_b1,      // [B, H] bf16 ping-pong
               float* c_io,                     // [B, H] f32 (col-partitioned)
               float* __restrict__ out,         // [T, B, H]
               unsigned* barrier_ctr) {
  __shared__ uint4 smemB[64 * LDS_ROW_U4];     // 132,096 B staged W_hh slice
  __shared__ float smemP[4 * 4 * 256];         //  16,384 B gate partials

  const int tid  = threadIdx.x;
  const int lane = tid & 31;
  const int wave = tid >> 5;          // 0..7
  const int mt   = wave >> 1;         // m-tile 0..3
  const int kh   = wave & 1;          // k-half 0..1
  const int n0   = blockIdx.x * 16;

  // ---- stage W_hh rows {g*HID + n0 + 0..15 : g=0..3} into LDS ----
  for (int i = tid; i < 64 * 128; i += 256) {   // 64 rows x 128 uint4/row
    const int r = i >> 7;
    const int c = i & 127;
    const int gr = (r >> 4) * HID + n0 + (r & 15);
    async_cp16(smemB + (size_t)r * LDS_ROW_U4 + c,
               Whh + (size_t)gr * KDIM + (c << 3));
  }
  wait_async_all();
  __syncthreads();

  const int    kbase = kh * 512;
  const int    ncol  = n0 + (lane & 15);
  const size_t arow  = (size_t)(mt * 16 + (lane & 15));
  const int    rbase = mt * 16 + ((lane >> 4) << 3);
  const float  rr    = ret[ncol];
  const int    br0 = 0 * 16 + (lane & 15);   // LDS rows of the 4 gate columns
  const int    br1 = 1 * 16 + (lane & 15);
  const int    br2 = 2 * 16 + (lane & 15);
  const int    br3 = 3 * 16 + (lane & 15);

  unsigned gen = 0;
  for (int t = 0; t < T_STEPS; ++t) {
    const int cur = t & 1;
    const __bf16* hb   = cur ? h_b1 : h_b0;
    const float*  hf   = cur ? h_f1 : h_f0;
    float*        hf_o = cur ? h_f0 : h_f1;
    __bf16*       hb_o = cur ? h_b0 : h_b1;
    const __bf16* XGt  = XG + (size_t)t * BATCH * G4;

    // ---- partial GEMM over this wave's K half ----
    v8f acc0 = {}, acc1 = {}, acc2 = {}, acc3 = {};
    for (int kk = 0; kk < 512; kk += 32) {
      const int k = kbase + kk;
      v16bf a  = load_a_frag(hb, arow, k, lane);
      v16bf b0 = lds_b_frag(smemB, br0, k, lane);
      v16bf b1 = lds_b_frag(smemB, br1, k, lane);
      v16bf b2 = lds_b_frag(smemB, br2, k, lane);
      v16bf b3 = lds_b_frag(smemB, br3, k, lane);
      acc0 = __builtin_amdgcn_wmma_f32_16x16x32_bf16(false, a, false, b0, (short)0, acc0, false, false);
      acc1 = __builtin_amdgcn_wmma_f32_16x16x32_bf16(false, a, false, b1, (short)0, acc1, false, false);
      acc2 = __builtin_amdgcn_wmma_f32_16x16x32_bf16(false, a, false, b2, (short)0, acc2, false, false);
      acc3 = __builtin_amdgcn_wmma_f32_16x16x32_bf16(false, a, false, b3, (short)0, acc3, false, false);
    }

    // ---- k-half 1 dumps partials ----
    if (kh == 1) {
#pragma unroll
      for (int v = 0; v < 8; ++v) {
        smemP[((mt * 4 + 0) << 8) + (v << 5) + lane] = acc0[v];
        smemP[((mt * 4 + 1) << 8) + (v << 5) + lane] = acc1[v];
        smemP[((mt * 4 + 2) << 8) + (v << 5) + lane] = acc2[v];
        smemP[((mt * 4 + 3) << 8) + (v << 5) + lane] = acc3[v];
      }
    }
    __syncthreads();

    // ---- k-half 0 reduces + full elementwise LSTM / low-pass blend ----
    if (kh == 0) {
#pragma unroll
      for (int v = 0; v < 8; ++v) {
        const int row = rbase + v;
        const size_t gbase = (size_t)row * G4 + ncol;
        const size_t hidx  = (size_t)row * HID + ncol;
        float gi = acc0[v] + smemP[((mt * 4 + 0) << 8) + (v << 5) + lane] + (float)XGt[gbase + 0 * HID];
        float gf = acc1[v] + smemP[((mt * 4 + 1) << 8) + (v << 5) + lane] + (float)XGt[gbase + 1 * HID];
        float gg = acc2[v] + smemP[((mt * 4 + 2) << 8) + (v << 5) + lane] + (float)XGt[gbase + 2 * HID];
        float go = acc3[v] + smemP[((mt * 4 + 3) << 8) + (v << 5) + lane] + (float)XGt[gbase + 3 * HID];

        float cy = sigf(gf) * c_io[hidx] + sigf(gi) * tanhf(gg);
        float hy = sigf(go) * tanhf(cy);
        float hbv = rr * hf[hidx] + (1.0f - rr) * hy;

        c_io[hidx]  = cy;
        hf_o[hidx]  = hbv;
        hb_o[hidx]  = (__bf16)hbv;
        out[(size_t)t * BATCH * HID + hidx] = hbv;
      }
    }
    __syncthreads();            // partials consumed; block step done

    // ---- device-wide generation barrier (release h, acquire others' h) ----
    __threadfence();            // flush this WGP's h stores to device scope
    gen++;
    if (tid == 0) {
      __hip_atomic_fetch_add(barrier_ctr, 1u, __ATOMIC_ACQ_REL,
                             __HIP_MEMORY_SCOPE_AGENT);
      const unsigned target = gen * (unsigned)NBLK;
      while (__hip_atomic_load(barrier_ctr, __ATOMIC_ACQUIRE,
                               __HIP_MEMORY_SCOPE_AGENT) < target) {
        __builtin_amdgcn_s_sleep(2);
      }
    }
    __syncthreads();
    __threadfence();            // invalidate so next step's h loads see L2
  }
}

// ---------------------------------------------------------------------------
// Host launcher
// ---------------------------------------------------------------------------
extern "C" void kernel_launch(void* const* d_in, const int* in_sizes, int n_in,
                              void* d_out, int out_size, void* d_ws, size_t ws_size,
                              hipStream_t stream) {
  (void)in_sizes; (void)n_in; (void)out_size; (void)ws_size;

  const float* input_ = (const float*)d_in[0];  // [T, B, I]
  const float* h0     = (const float*)d_in[1];  // [B, H]
  const float* c0     = (const float*)d_in[2];  // [B, H]
  const float* w_ih   = (const float*)d_in[3];  // [4H, I]
  const float* w_hh   = (const float*)d_in[4];  // [4H, H]
  const float* b_ih   = (const float*)d_in[5];  // [4H]
  const float* b_hh   = (const float*)d_in[6];  // [4H]
  const float* ret    = (const float*)d_in[7];  // [H]
  float* out = (float*)d_out;                   // outputs [T,B,H] ++ hT ++ cT

  // ---- workspace carve-out (bytes) ----
  char* ws = (char*)d_ws;
  const size_t XG_BYTES  = (size_t)T_STEPS * BATCH * G4 * 2;   // 256 MB bf16
  const size_t XBF_BYTES = (size_t)T_STEPS * BATCH * KDIM * 2; //  64 MB bf16
  const size_t W_BYTES   = (size_t)G4 * KDIM * 2;              //   8 MB bf16
  size_t off = 0;
  __bf16* xg     = (__bf16*)(ws + off); off += XG_BYTES;
  __bf16* x_bf   = (__bf16*)(ws + off); off += XBF_BYTES;
  __bf16* wih_bf = (__bf16*)(ws + off); off += W_BYTES;
  __bf16* whh_bf = (__bf16*)(ws + off); off += W_BYTES;
  float*  h_f0   = (float*) (ws + off); off += (size_t)BATCH * HID * 4;
  float*  h_f1   = (float*) (ws + off); off += (size_t)BATCH * HID * 4;
  __bf16* h_b0   = (__bf16*)(ws + off); off += (size_t)BATCH * HID * 2;
  __bf16* h_b1   = (__bf16*)(ws + off); off += (size_t)BATCH * HID * 2;
  float*  c_f    = (float*) (ws + off); off += (size_t)BATCH * HID * 4;
  unsigned* ctr  = (unsigned*)(ws + off); off += 256;

  // ---- 1. convert inputs / weights to bf16 ----
  const long long NX = (long long)T_STEPS * BATCH * KDIM;
  const long long NW = (long long)G4 * KDIM;
  cvt_f32_bf16<<<16384, 256, 0, stream>>>(input_, x_bf, NX);
  cvt_f32_bf16<<<8192, 256, 0, stream>>>(w_ih, wih_bf, NW);
  cvt_f32_bf16<<<8192, 256, 0, stream>>>(w_hh, whh_bf, NW);

  // ---- 2. state + barrier init ----
  init_state<<<(BATCH * HID) / 256, 256, 0, stream>>>(h0, c0, h_f0, h_b0, c_f);
  (void)hipMemsetAsync(ctr, 0, sizeof(unsigned), stream);

  // ---- 3. x_gates = X @ Wih^T + b_ih + b_hh (bf16 WMMA, fp32 accum) ----
  xgates_gemm<<<dim3(256, 64), 256, 0, stream>>>(x_bf, wih_bf, b_ih, b_hh, xg);

  // ---- 4. persistent recurrence: single kernel, 512 internal steps ----
  lstm_scan<<<NBLK, 256, 0, stream>>>(xg, whh_bf, ret,
                                      h_f0, h_f1, h_b0, h_b1, c_f,
                                      out, ctr);

  // ---- 5. final hT, cT (after 512 steps the live h buffer is index 0) ----
  const size_t tail = (size_t)T_STEPS * BATCH * HID;
  (void)hipMemcpyAsync(out + tail, h_f0, (size_t)BATCH * HID * 4,
                       hipMemcpyDeviceToDevice, stream);
  (void)hipMemcpyAsync(out + tail + (size_t)BATCH * HID, c_f, (size_t)BATCH * HID * 4,
                       hipMemcpyDeviceToDevice, stream);
}
